// DemopackDecoder_70274254897207
// MI455X (gfx1250) — matrix operand order repaired and checked by reading, not compiled, not run on hardware
//
#include <hip/hip_runtime.h>

// ---------------- types ----------------
typedef __bf16 bf16_t;
typedef bf16_t v16bf __attribute__((ext_vector_type(16)));
typedef float  v8f   __attribute__((ext_vector_type(8)));

struct alignas(16) F4  { float x, y, z, w; };
struct alignas(8)  US4 { unsigned short x, y, z, w; };

union AB16 { v16bf v; F4 f[2]; };   // 16 bf16 = 32 bytes = two 16B chunks

__device__ __forceinline__ unsigned short f2bf(float f) {
    // native RTNE f32 -> bf16 (backend selects hw convert on gfx1250)
    __bf16 h = (__bf16)f;
    return __builtin_bit_cast(unsigned short, h);
}

// ---------------- kernel 1: codebook gather + mean -> bf16 ----------------
// one block per output row (T*R = 4096 rows), D = 4096
__global__ void __launch_bounds__(256) decode_kernel(
    const float* __restrict__ cw,          // [16384, 4096]
    const int*   __restrict__ idx,         // [T*R*4]
    unsigned short* __restrict__ decoded)  // [T*R, 4096] bf16 bits
{
    const int D = 4096;
    const int row = blockIdx.x;
    const int tid = threadIdx.x;

    const float* c0 = cw + (size_t)idx[row * 4 + 0] * D;
    const float* c1 = cw + (size_t)idx[row * 4 + 1] * D;
    const float* c2 = cw + (size_t)idx[row * 4 + 2] * D;
    const float* c3 = cw + (size_t)idx[row * 4 + 3] * D;
    unsigned short* dst = decoded + (size_t)row * D;

#pragma unroll
    for (int c = 0; c < 4; ++c) {
        const int d = c * 1024 + tid * 4;      // coalesced float4 sweeps
        F4 a = *(const F4*)(c0 + d);
        F4 b = *(const F4*)(c1 + d);
        F4 e = *(const F4*)(c2 + d);
        F4 g = *(const F4*)(c3 + d);
        US4 o;
        o.x = f2bf((a.x + b.x + e.x + g.x) * 0.25f);
        o.y = f2bf((a.y + b.y + e.y + g.y) * 0.25f);
        o.z = f2bf((a.z + b.z + e.z + g.z) * 0.25f);
        o.w = f2bf((a.w + b.w + e.w + g.w) * 0.25f);
        *(US4*)(dst + d) = o;
    }
}

// ---------------- kernel 2: Y[t] = x @ rot[t]^T, bf16 out -----------------
// Y[t][n_tok][d] = sum_e x[n_tok][e] * rot[t][d][e]
// A = x (M=512 tokens, K=e), B col n=d is rot row d (contiguous along K) ->
// both stagings are straight row copies (fp32 -> bf16 convert, no transpose).
// grid: (N/128=32, M/256=2, T=8), block 256 = 8 waves, wave tile 64x64, BK=32
__global__ void __launch_bounds__(256) rot_apply_kernel(
    const float* __restrict__ x,    // [512, 4096] f32
    const float* __restrict__ rot,  // [T, 4096, 4096] f32
    unsigned short* __restrict__ Y) // [T, 512, 4096] bf16
{
    const int D = 4096;
    __shared__ unsigned short As[256 * 32];  // [m][k], 16KB
    __shared__ unsigned short Bs[128 * 32];  // [n][k],  8KB

    const int t  = blockIdx.z;
    const int n0 = blockIdx.x * 128;
    const int m0 = blockIdx.y * 256;
    const float* Bg = rot + (size_t)t * D * D;

    const int tid  = threadIdx.x;
    const int lane = tid & 31;
    const int wid  = tid >> 5;
    const int wm   = (wid & 3) * 64;   // 4 waves along M
    const int wn   = (wid >> 2) * 64;  // 2 waves along N
    const int lr   = lane & 15;
    const int lh   = lane >> 4;

    // staging: 8 threads per row, 4 floats each -> 128B contiguous per 8 lanes
    const int r8 = tid >> 3;          // 0..31
    const int c4 = (tid & 7) * 4;     // 0,4,...,28

    v8f acc[4][4] = {};

    for (int k0 = 0; k0 < D; k0 += 32) {
        // A tile: x rows m0..m0+255 (coalesced)
#pragma unroll
        for (int c = 0; c < 8; ++c) {
            const int row = r8 + c * 32;
            F4 v = *(const F4*)(x + (size_t)(m0 + row) * D + k0 + c4);
            US4 o = { f2bf(v.x), f2bf(v.y), f2bf(v.z), f2bf(v.w) };
            *(US4*)(As + row * 32 + c4) = o;
        }
        // B tile: rot rows n0..n0+127 (coalesced)
#pragma unroll
        for (int c = 0; c < 4; ++c) {
            const int row = r8 + c * 32;
            F4 v = *(const F4*)(Bg + (size_t)(n0 + row) * D + k0 + c4);
            US4 o = { f2bf(v.x), f2bf(v.y), f2bf(v.z), f2bf(v.w) };
            *(US4*)(Bs + row * 32 + c4) = o;
        }
        __syncthreads();

        AB16 a[4], b[4];
#pragma unroll
        for (int mi = 0; mi < 4; ++mi) {
            // A 16x32 layout: lane<16 -> K 0..7,16..23 ; lane>=16 -> K 8..15,24..31
            const unsigned short* p = As + (wm + mi * 16 + lr) * 32 + lh * 8;
            a[mi].f[0] = *(const F4*)(p);
            a[mi].f[1] = *(const F4*)(p + 16);
        }
#pragma unroll
        for (int ni = 0; ni < 4; ++ni) {
            // B 32x16 layout: lane<16 -> col n, K 0..7,16..23 ; lane>=16 -> K 8..15,24..31
            const unsigned short* p = Bs + (wn + ni * 16 + lr) * 32 + lh * 8;
            b[ni].f[0] = *(const F4*)(p);
            b[ni].f[1] = *(const F4*)(p + 16);
        }
#pragma unroll
        for (int mi = 0; mi < 4; ++mi)
#pragma unroll
            for (int ni = 0; ni < 4; ++ni)
                acc[mi][ni] = __builtin_amdgcn_wmma_f32_16x16x32_bf16(
                    false, a[mi].v, false, b[ni].v, (short)0, acc[mi][ni],
                    false, false);
        __syncthreads();
    }

    unsigned short* Yt = Y + (size_t)t * 512 * D;
#pragma unroll
    for (int mi = 0; mi < 4; ++mi)
#pragma unroll
        for (int ni = 0; ni < 4; ++ni) {
            const int col = n0 + wn + ni * 16 + lr;
#pragma unroll
            for (int v = 0; v < 8; ++v) {
                const int mrow = m0 + wm + mi * 16 + lh * 8 + v;
                Yt[(size_t)mrow * D + col] = f2bf(acc[mi][ni][v]);
            }
        }
}

// ---------------- kernel 3: out[:, t*512+r] = s[t]*(Y[t] @ decoded[t]^T) + bias
// A = Y[t] (bf16), B col n=r is decoded row r (contiguous along K=d):
// both operands are bf16 in final LDS layout -> async DMA straight to LDS.
// grid: (N/128=4, M/128=4, T=8), block 256 = 8 waves, wave tile 32x64
__global__ void __launch_bounds__(256) out_gemm_kernel(
    const unsigned short* __restrict__ Y,       // [T, 512, 4096] bf16
    const unsigned short* __restrict__ decoded, // [T*512, 4096] bf16
    const float*          __restrict__ scales,  // [T]
    const float*          __restrict__ bias,    // [4096]
    float*                __restrict__ out)     // [512, 4096] f32
{
    const int D = 4096;
    __shared__ unsigned short As[128 * 32];  // [m][k]
    __shared__ unsigned short Bs[128 * 32];  // [n][k]

    const int t  = blockIdx.z;
    const int n0 = blockIdx.x * 128;   // r within tile (0..511)
    const int m0 = blockIdx.y * 128;   // token
    const unsigned short* Yt = Y       + (size_t)t * 512 * D;
    const unsigned short* Dt = decoded + (size_t)t * 512 * D;

    const int tid  = threadIdx.x;
    const int lane = tid & 31;
    const int wid  = tid >> 5;
    const int wm   = (wid & 3) * 32;
    const int wn   = (wid >> 2) * 64;
    const int lr   = lane & 15;
    const int lh   = lane >> 4;

    const int s_row = tid >> 1;
    const int s_seg = (tid & 1) * 16;

    v8f acc[2][4] = {};

    for (int k0 = 0; k0 < D; k0 += 32) {
        // async memory->LDS DMA (ASYNCcnt), 16B per lane per op
        {
            const unsigned short* sa = Yt + (size_t)(m0 + s_row) * D + k0 + s_seg;
            const unsigned short* sb = Dt + (size_t)(n0 + s_row) * D + k0 + s_seg;
            unsigned dA = (unsigned)(size_t)(As + s_row * 32 + s_seg);
            unsigned dB = (unsigned)(size_t)(Bs + s_row * 32 + s_seg);
            asm volatile("global_load_async_to_lds_b128 %0, %1, off"
                         :: "v"(dA),      "v"(sa)     : "memory");
            asm volatile("global_load_async_to_lds_b128 %0, %1, off"
                         :: "v"(dA + 16), "v"(sa + 8) : "memory");
            asm volatile("global_load_async_to_lds_b128 %0, %1, off"
                         :: "v"(dB),      "v"(sb)     : "memory");
            asm volatile("global_load_async_to_lds_b128 %0, %1, off"
                         :: "v"(dB + 16), "v"(sb + 8) : "memory");
            asm volatile("s_wait_asynccnt 0" ::: "memory");
        }
        __syncthreads();

        AB16 a[2], b[4];
#pragma unroll
        for (int mi = 0; mi < 2; ++mi) {
            const unsigned short* p = As + (wm + mi * 16 + lr) * 32 + lh * 8;
            a[mi].f[0] = *(const F4*)(p);
            a[mi].f[1] = *(const F4*)(p + 16);
        }
#pragma unroll
        for (int ni = 0; ni < 4; ++ni) {
            const unsigned short* p = Bs + (wn + ni * 16 + lr) * 32 + lh * 8;
            b[ni].f[0] = *(const F4*)(p);
            b[ni].f[1] = *(const F4*)(p + 16);
        }
#pragma unroll
        for (int mi = 0; mi < 2; ++mi)
#pragma unroll
            for (int ni = 0; ni < 4; ++ni)
                acc[mi][ni] = __builtin_amdgcn_wmma_f32_16x16x32_bf16(
                    false, a[mi].v, false, b[ni].v, (short)0, acc[mi][ni],
                    false, false);
        __syncthreads();
    }

    const float s = scales[t];
#pragma unroll
    for (int mi = 0; mi < 2; ++mi)
#pragma unroll
        for (int ni = 0; ni < 4; ++ni) {
            const int col = t * 512 + n0 + wn + ni * 16 + lr;   // out feature
            const float bv = bias[col];
#pragma unroll
            for (int v = 0; v < 8; ++v) {
                const int mrow = m0 + wm + mi * 16 + lh * 8 + v;  // token
                out[(size_t)mrow * D + col] = s * acc[mi][ni][v] + bv;
            }
        }
}

// ---------------- launch ----------------
extern "C" void kernel_launch(void* const* d_in, const int* in_sizes, int n_in,
                              void* d_out, int out_size, void* d_ws, size_t ws_size,
                              hipStream_t stream) {
    const float* x       = (const float*)d_in[0];
    const float* cw      = (const float*)d_in[1];
    const int*   idx     = (const int*)  d_in[2];
    const float* rot     = (const float*)d_in[3];
    const float* scales  = (const float*)d_in[4];
    const float* bias    = (const float*)d_in[5];
    float*       out     = (float*)d_out;

    unsigned short* decoded = (unsigned short*)d_ws;           // 32MB
    unsigned short* Y       = decoded + (size_t)4096 * 4096;   // 32MB

    // 1) gather + mean -> bf16 decoded rows (memory bound, ~256MB gather)
    decode_kernel<<<4096, 256, 0, stream>>>(cw, idx, decoded);

    // 2) Y[t] = x @ rot[t]^T  (137 GFLOP, rot streamed ~once from HBM)
    dim3 gB(32, 2, 8);
    rot_apply_kernel<<<gB, 256, 0, stream>>>(x, rot, Y);

    // 3) out = s[t] * Y[t] @ decoded[t]^T + bias (17 GFLOP, L2-resident)
    dim3 gC(4, 4, 8);
    out_gemm_kernel<<<gC, 256, 0, stream>>>(Y, decoded, scales, bias, out);
}
